// CausalSelfAttention_78254304133326
// MI455X (gfx1250) — matrix operand order repaired and checked by reading, not compiled, not run on hardware
//
#include <hip/hip_runtime.h>

// ---------------------------------------------------------------------------
// Causal self-attention for MI455X (gfx1250), bf16 WMMA with fp32 accumulate.
// B=4, S=2048, D=P=1024. All matrix math via v_wmma_f32_16x16x32_bf16.
// Hot loop: uniform SGPR base pointers advance by the K step; per-lane 32-bit
// voffsets are loop-invariant; chunk selection uses immediate offsets. This
// keeps the loop body to {global_load_b128, v_wmma} with no VALU address math
// (and therefore no WMMA->VALU hazard v_nops).
// ---------------------------------------------------------------------------

typedef unsigned short u16;
typedef __attribute__((ext_vector_type(16))) __bf16 v16bf;
typedef __attribute__((ext_vector_type(8)))  __bf16 v8bf;
typedef __attribute__((ext_vector_type(8)))  float  v8f;

static __device__ __forceinline__ u16 f32_to_bf16_bits(float f) {
    union { float f; unsigned u; } c; c.f = f;
    unsigned u = c.u;
    if ((u & 0x7fffffffu) > 0x7f800000u) return (u16)((u >> 16) | 0x0040u); // quiet NaN
    unsigned lsb = (u >> 16) & 1u;
    return (u16)((u + 0x7fffu + lsb) >> 16);                                // RNE
}

// ------------------------------ conversions --------------------------------

__global__ __launch_bounds__(256) void cvt_f32_bf16(const float* __restrict__ in,
                                                    u16* __restrict__ out, int n) {
    int i = blockIdx.x * blockDim.x + threadIdx.x;
    if (i < n) out[i] = f32_to_bf16_bits(in[i]);
}

// W: R x C row-major (fp32)  ->  Wt: C x R (bf16 bits), i.e. n-major for GEMM B.
__global__ __launch_bounds__(256) void transpose_cvt(const float* __restrict__ W,
                                                     u16* __restrict__ Wt, int R, int C) {
    int i = blockIdx.x * blockDim.x + threadIdx.x;
    if (i >= R * C) return;
    int r = i / C, c = i % C;
    Wt[(size_t)c * R + r] = f32_to_bf16_bits(W[i]);
}

// ------------------------------ WMMA GEMM ----------------------------------
// C = A (M x K, row-major bf16) * Bt^T, Bt stored n-major (N x K row-major)
// so every B fragment is one contiguous 32B chunk along K.
//
// Block tile 128(M) x 256(N); 8 wave32 waves arranged 2(M) x 4(N); each wave
// owns a 64x64 tile = 4x4 WMMA subtiles (16 x v8f accumulators).
//
// EPI: 0 = bf16 row-major out        (Q, K projections)
//      1 = bf16 per-batch transposed (V projection -> (B,P,S))
//      2 = fp32 out, scaled          (attention scores, causal tile skip)
//      3 = fp32 out                  (final context)
// CAUSAL_TILE: skip output tiles entirely above the diagonal (scores).
// CAUSAL_K:    truncate the K loop at the diagonal (context; probs==0 beyond).

template <int EPI, bool CAUSAL_TILE, bool CAUSAL_K>
__global__ __launch_bounds__(256) void gemm_bf16_wmma(
    const u16* __restrict__ A, const u16* __restrict__ Bt, void* __restrict__ Cv,
    int M, int N, int K,
    size_t sA, size_t sB, size_t sC,       // per-batch element strides
    float scale, int rows_per_batch)
{
    constexpr int BM = 128, BN = 256;
    const int m0 = blockIdx.y * BM;
    const int n0 = blockIdx.x * BN;
    if (CAUSAL_TILE && n0 >= m0 + BM) return;   // tile fully above diagonal

    const int b = blockIdx.z;

    const int lane = threadIdx.x & 31;
    const int wid  = threadIdx.x >> 5;
    const int wm   = wid & 1;           // 2 wave rows  (64 each)
    const int wn   = wid >> 1;          // 4 wave cols  (64 each)
    const int h    = lane >> 4;         // half-wave
    const int l16  = lane & 15;

    const int mw = m0 + wm * 64;        // wave row base (4 subtiles of 16)
    const int nw = n0 + wn * 64;        // wave col base (4 subtiles of 16)

    int Keff = K;
    if (CAUSAL_K) { int kl = m0 + BM; Keff = kl < K ? kl : K; }

    // Loop-invariant per-lane byte offsets; uniform bases advance by 64B/step.
    unsigned aoff[4], boff[4];
#pragma unroll
    for (int i = 0; i < 4; ++i)
        aoff[i] = ((unsigned)(mw + i * 16 + l16) * (unsigned)K + 16u * h) * 2u;
#pragma unroll
    for (int j = 0; j < 4; ++j)
        boff[j] = ((unsigned)(nw + j * 16 + l16) * (unsigned)K + 16u * h) * 2u;

    const char* Ak = (const char*)(A + (size_t)b * sA);
    const char* Bk = (const char*)(Bt + (size_t)b * sB);

    v8f acc[4][4] = {};

    for (int kk = 0; kk < Keff; kk += 32) {
        v16bf bfr[4];
#pragma unroll
        for (int j = 0; j < 4; ++j)     // B frag: one 32B chunk at k+16h
            bfr[j] = *(const v16bf*)(Bk + boff[j]);
#pragma unroll
        for (int i = 0; i < 4; ++i) {   // A frag: two 16B chunks (k+16h, k+16h+8)
            v8bf lo = *(const v8bf*)(Ak + aoff[i]);
            v8bf hi = *(const v8bf*)(Ak + aoff[i] + 16);
            v16bf af;
#pragma unroll
            for (int e = 0; e < 8; ++e) { af[e] = lo[e]; af[8 + e] = hi[e]; }
#pragma unroll
            for (int j = 0; j < 4; ++j)
                acc[i][j] = __builtin_amdgcn_wmma_f32_16x16x32_bf16(
                    false, af, false, bfr[j], (short)0, acc[i][j], false, false);
        }
        Ak += 64;                       // 32 bf16 along K
        Bk += 64;
    }

    // ---------------- epilogue: C/D layout row = r + 8h, col = l16 ----------
    const int bb    = (EPI == 1) ? (m0 / rows_per_batch) : 0;
    const int tbase = (EPI == 1) ? (m0 % rows_per_batch) : 0;
#pragma unroll
    for (int i = 0; i < 4; ++i) {
#pragma unroll
        for (int j = 0; j < 4; ++j) {
#pragma unroll
            for (int r = 0; r < 8; ++r) {
                const int m = mw + i * 16 + r + 8 * h;
                const int n = nw + j * 16 + l16;
                const float v = acc[i][j][r];
                if (EPI == 0) {
                    u16* C = (u16*)Cv + (size_t)b * sC;
                    C[(size_t)m * N + n] = f32_to_bf16_bits(v);
                } else if (EPI == 1) {
                    u16* C = (u16*)Cv;
                    const int t = tbase + (m - m0);
                    C[((size_t)bb * N + n) * rows_per_batch + t] = f32_to_bf16_bits(v);
                } else if (EPI == 2) {
                    float* C = (float*)Cv + (size_t)b * sC;
                    C[(size_t)m * N + n] = v * scale;
                } else {
                    float* C = (float*)Cv + (size_t)b * sC;
                    C[(size_t)m * N + n] = v;
                }
            }
        }
    }
}

// ------------------------------ softmax ------------------------------------
// One 256-thread block per (row, batch). Causal length = row+1; writes bf16
// probabilities, explicit zeros for t > row (context GEMM reads t up to the last row of its 128-row tile).

__global__ __launch_bounds__(256) void softmax_causal(const float* __restrict__ scores,
                                                      u16* __restrict__ probs, int S) {
    const int row = blockIdx.x, b = blockIdx.y;
    const float* src = scores + ((size_t)b * S + row) * S;
    u16* dst = probs + ((size_t)b * S + row) * S;
    const int n = row + 1;
    const int tid = threadIdx.x;
    __shared__ float red[256];

    float m = -3.4e38f;
    for (int i = tid; i < n; i += 256) m = fmaxf(m, src[i]);
    red[tid] = m; __syncthreads();
    for (int s = 128; s > 0; s >>= 1) {
        if (tid < s) red[tid] = fmaxf(red[tid], red[tid + s]);
        __syncthreads();
    }
    m = red[0]; __syncthreads();

    float sum = 0.f;
    for (int i = tid; i < n; i += 256) sum += __expf(src[i] - m);
    red[tid] = sum; __syncthreads();
    for (int s = 128; s > 0; s >>= 1) {
        if (tid < s) red[tid] += red[tid + s];
        __syncthreads();
    }
    const float inv = 1.0f / red[0];

    for (int i = tid; i < S; i += 256) {
        const float p = (i < n) ? __expf(src[i] - m) * inv : 0.0f;
        dst[i] = f32_to_bf16_bits(p);
    }
}

// ------------------------------ launcher -----------------------------------

extern "C" void kernel_launch(void* const* d_in, const int* in_sizes, int n_in,
                              void* d_out, int out_size, void* d_ws, size_t ws_size,
                              hipStream_t stream)
{
    (void)in_sizes; (void)n_in; (void)out_size; (void)ws_size;
    const int B = 4, S = 2048, D = 1024, P = 1024;
    const int MS = B * S;                       // 8192 total rows

    const float* x  = (const float*)d_in[0];
    const float* Wq = (const float*)d_in[1];
    const float* Wk = (const float*)d_in[2];
    const float* Wv = (const float*)d_in[3];
    float* out = (float*)d_out;

    // Workspace carve-out (~166 MiB total).
    char* ws = (char*)d_ws;
    size_t off = 0;
    auto alloc = [&](size_t bytes) -> void* {
        void* p = ws + off;
        off += (bytes + 255) & ~(size_t)255;
        return p;
    };
    u16*   xb  = (u16*)alloc((size_t)MS * D * 2);      // x in bf16
    u16*   WtQ = (u16*)alloc((size_t)D * P * 2);       // Wq^T bf16 (n-major)
    u16*   WtK = (u16*)alloc((size_t)D * P * 2);
    u16*   WtV = (u16*)alloc((size_t)D * P * 2);
    u16*   Qb  = (u16*)alloc((size_t)MS * P * 2);      // Q bf16 (B*S, P)
    u16*   Kb  = (u16*)alloc((size_t)MS * P * 2);      // K bf16 (B*S, P)
    u16*   Vt  = (u16*)alloc((size_t)MS * P * 2);      // V^T bf16 (B, P, S)
    float* sc  = (float*)alloc((size_t)B * S * S * 4); // scores fp32
    u16*   pr  = (u16*)alloc((size_t)B * S * S * 2);   // probs bf16

    // 1) conversions
    { int n = MS * D; cvt_f32_bf16<<<(n + 255) / 256, 256, 0, stream>>>(x, xb, n); }
    { int n = D * P;
      transpose_cvt<<<(n + 255) / 256, 256, 0, stream>>>(Wq, WtQ, D, P);
      transpose_cvt<<<(n + 255) / 256, 256, 0, stream>>>(Wk, WtK, D, P);
      transpose_cvt<<<(n + 255) / 256, 256, 0, stream>>>(Wv, WtV, D, P); }

    const dim3 blk(256);

    // 2) projections: (8192 x 1024) = (8192 x 1024) x (1024 x 1024)
    {
        dim3 grid(P / 256, MS / 128, 1);
        gemm_bf16_wmma<0, false, false><<<grid, blk, 0, stream>>>(
            xb, WtQ, Qb, MS, P, D, 0, 0, 0, 1.f, S);
        gemm_bf16_wmma<0, false, false><<<grid, blk, 0, stream>>>(
            xb, WtK, Kb, MS, P, D, 0, 0, 0, 1.f, S);
        gemm_bf16_wmma<1, false, false><<<grid, blk, 0, stream>>>(
            xb, WtV, Vt, MS, P, D, 0, 0, 0, 1.f, S);
    }

    // 3) scores = Q K^T / sqrt(P), causal tile skip. Per batch: 2048x2048x1024.
    {
        dim3 grid(S / 256, S / 128, B);
        gemm_bf16_wmma<2, true, false><<<grid, blk, 0, stream>>>(
            Qb, Kb, sc, S, S, P,
            (size_t)S * P, (size_t)S * P, (size_t)S * S, 1.0f / 32.0f, S);
    }

    // 4) causal softmax -> bf16 probs (zeros beyond diagonal)
    {
        dim3 grid(S, B);
        softmax_causal<<<grid, blk, 0, stream>>>(sc, pr, S);
    }

    // 5) context = probs @ V, K loop truncated at diagonal. Per batch: 2048x1024x2048.
    {
        dim3 grid(P / 256, S / 128, B);
        gemm_bf16_wmma<3, false, true><<<grid, blk, 0, stream>>>(
            pr, Vt, out, S, P, S,
            (size_t)S * S, (size_t)P * S, (size_t)S * P, 1.f, S);
    }
}
